// FeatureMatching_10453950398626
// MI455X (gfx1250) — matrix operand order repaired and checked by reading, not compiled, not run on hardware
//
#include <hip/hip_runtime.h>
#include <hip/hip_bf16.h>

typedef __attribute__((ext_vector_type(16))) _Float16 v16h;
typedef __attribute__((ext_vector_type(8)))  _Float16 v8h;
typedef __attribute__((ext_vector_type(8)))  float    v8f;

#define B_     4
#define C_     64
#define H_     64
#define W_     64
#define L_     4096          // H*W
#define D_     576           // C*9
#define KSTEPS 18            // D / 32
#define KSPLIT 4
#define KPER   (L_ / KSPLIT) // 1024 keys per split
#define KTILES (KPER / 16)   // 64 tiles of 16 keys

__device__ __forceinline__ int refl(int i) {
    // jnp 'reflect' (no edge repeat): p[-1]=x[1], p[H]=x[H-2]
    return i < 0 ? -i : (i > 63 ? 126 - i : i);
}

// ---------------------------------------------------------------------------
// Kernel 1: patch extraction (3x3 reflect) + L2 normalization -> f16 rows
// grid: (L_, B_, 2), block: 64 threads (one per channel)
// dst layout: [L_, D_] row-major, d = c*9 + kh*3 + kw (torch Unfold order)
// ---------------------------------------------------------------------------
__global__ void fm_patchify(const float* __restrict__ query,
                            const float* __restrict__ key,
                            _Float16* __restrict__ qn,
                            _Float16* __restrict__ kn) {
    const int p = blockIdx.x;            // pixel 0..4095
    const int b = blockIdx.y;            // batch
    const int t = blockIdx.z;            // 0 = query, 1 = key
    const int c = threadIdx.x;           // channel 0..63

    const float* src = (t == 0 ? query : key) + ((size_t)b * C_ + c) * (H_ * W_);
    _Float16* dst = (t == 0 ? qn : kn) + ((size_t)b * L_ + p) * D_ + c * 9;

    const int y = p >> 6, x = p & 63;
    float v[9];
    float ss = 0.0f;
#pragma unroll
    for (int kh = 0; kh < 3; ++kh) {
        const int yy = refl(y + kh - 1);
#pragma unroll
        for (int kw = 0; kw < 3; ++kw) {
            const int xx = refl(x + kw - 1);
            const float f = src[yy * W_ + xx];
            v[kh * 3 + kw] = f;
            ss += f * f;
        }
    }

    // reduce squared norm over the 64 channels (2 waves -> LDS tree)
    __shared__ float red[64];
    red[c] = ss;
    __syncthreads();
    for (int st = 32; st > 0; st >>= 1) {
        if (c < st) red[c] += red[c + st];
        __syncthreads();
    }
    const float inv = 1.0f / fmaxf(sqrtf(red[0]), 1e-12f);

#pragma unroll
    for (int j = 0; j < 9; ++j) dst[j] = (_Float16)(v[j] * inv);
}

// ---------------------------------------------------------------------------
// Kernel 2: cosine-similarity GEMM via v_wmma_f32_16x16x32_f16 + running
// max/argmax over a key range.  grid: (32 qblocks, KSPLIT, B_), block 256.
// Each wave: 16 queries (B-frags held in registers), streams 16-key A tiles.
// ---------------------------------------------------------------------------
__global__ void __launch_bounds__(256)
fm_match(const _Float16* __restrict__ qn,
         const _Float16* __restrict__ kn,
         float* __restrict__ pBest,
         int* __restrict__ pIdx) {
    const int qb = blockIdx.x;   // 0..31  (128 queries per workgroup)
    const int s  = blockIdx.y;   // key split 0..3
    const int b  = blockIdx.z;   // batch

    const _Float16* Qn = qn + (size_t)b * L_ * D_;
    const _Float16* Kn = kn + (size_t)b * L_ * D_;

    const int wave = threadIdx.x >> 5;
    const int lane = threadIdx.x & 31;
    const int n    = lane & 15;   // column (query) / A row within tile
    const int hsel = lane >> 4;   // half-wave selector

    const int q0 = qb * 128 + wave * 16;

    // Preload the 18 query B-fragments for this wave's 16 columns.
    // B layout (32x16 f16): lane n holds col q0+n, halves K = hsel*8 + {0..7} and 16 + hsel*8 + {0..7}.
    const _Float16* qrow = Qn + (size_t)(q0 + n) * D_;
    v16h qf[KSTEPS];
#pragma unroll
    for (int kk = 0; kk < KSTEPS; ++kk) {
        const v8h qlo = *(const v8h*)(qrow + kk * 32 + hsel * 8);
        const v8h qhi = *(const v8h*)(qrow + kk * 32 + 16 + hsel * 8);
        qf[kk] = __builtin_shufflevector(qlo, qhi,
            0, 1, 2, 3, 4, 5, 6, 7, 8, 9, 10, 11, 12, 13, 14, 15);
    }

    float best = -3.0e38f;
    int   bidx = 0;
    const int koff = s * KPER;

    for (int kt = 0; kt < KTILES; ++kt) {
        // A layout (16x32 f16): lane holds row m = lane&15;
        // halves K = hsel*8 + {0..7} and 16 + hsel*8 + {0..7}.
        const _Float16* arow = Kn + (size_t)(koff + kt * 16 + n) * D_;
        if (kt + 1 < KTILES)
            __builtin_prefetch(Kn + (size_t)(koff + (kt + 1) * 16 + n) * D_, 0, 0);

        v8f acc = {};
#pragma unroll
        for (int kk = 0; kk < KSTEPS; ++kk) {
            const v8h lo = *(const v8h*)(arow + kk * 32 + hsel * 8);
            const v8h hi = *(const v8h*)(arow + kk * 32 + 16 + hsel * 8);
            const v16h a = __builtin_shufflevector(lo, hi,
                0, 1, 2, 3, 4, 5, 6, 7, 8, 9, 10, 11, 12, 13, 14, 15);
            acc = __builtin_amdgcn_wmma_f32_16x16x32_f16(
                false, a, false, qf[kk], (short)0, acc, false, false);
        }

        // C layout: lane n (+/-16) holds col q0+n; VGPR v -> key row hsel*8+v.
        const int mbase = koff + kt * 16 + hsel * 8;
#pragma unroll
        for (int v = 0; v < 8; ++v) {
            const float sc = acc[v];
            const int   ki = mbase + v;
            if (sc > best) { best = sc; bidx = ki; }  // strict > keeps first max
        }
    }

    // Merge the two half-wave candidates for each query column.
    const float ob = __shfl_xor(best, 16, 32);
    const int   oi = __shfl_xor(bidx, 16, 32);
    if (ob > best || (ob == best && oi < bidx)) { best = ob; bidx = oi; }

    if (lane < 16) {
        const size_t o = ((size_t)(b * KSPLIT + s)) * L_ + (q0 + n);
        pBest[o] = best;
        pIdx[o]  = bidx;
    }
}

// ---------------------------------------------------------------------------
// Kernel 3: combine KSPLIT partials -> final relevance + hard indices
// ---------------------------------------------------------------------------
__global__ void fm_combine(const float* __restrict__ pBest,
                           const int* __restrict__ pIdx,
                           float* __restrict__ outRel,
                           int* __restrict__ outIdx) {
    const int tid = blockIdx.x * blockDim.x + threadIdx.x;  // b*L_ + q
    if (tid >= B_ * L_) return;
    const int b = tid >> 12, q = tid & 4095;
    float best = -3.0e38f;
    int   bidx = 0;
#pragma unroll
    for (int s = 0; s < KSPLIT; ++s) {
        const size_t o = ((size_t)(b * KSPLIT + s)) * L_ + q;
        const float v = pBest[o];
        if (v > best) { best = v; bidx = pIdx[o]; }  // ascending key ranges
    }
    outRel[tid] = best;
    outIdx[tid] = bidx;
}

extern "C" void kernel_launch(void* const* d_in, const int* in_sizes, int n_in,
                              void* d_out, int out_size, void* d_ws, size_t ws_size,
                              hipStream_t stream) {
    (void)in_sizes; (void)n_in; (void)out_size; (void)ws_size;
    const float* query = (const float*)d_in[0];
    const float* key   = (const float*)d_in[1];

    // workspace layout
    _Float16* qn = (_Float16*)d_ws;                       // B_*L_*D_ halves
    _Float16* kn = qn + (size_t)B_ * L_ * D_;             // B_*L_*D_ halves
    float* pBest = (float*)(kn + (size_t)B_ * L_ * D_);   // B_*KSPLIT*L_
    int*   pIdx  = (int*)(pBest + (size_t)B_ * KSPLIT * L_);

    float* outRel = (float*)d_out;                        // [B_,1,H_,W_]
    int*   outIdx = (int*)d_out + (size_t)B_ * L_;        // [B_, L_] int32

    fm_patchify<<<dim3(L_, B_, 2), 64, 0, stream>>>(query, key, qn, kn);
    fm_match<<<dim3(32, KSPLIT, B_), 256, 0, stream>>>(qn, kn, pBest, pIdx);
    fm_combine<<<(B_ * L_ + 255) / 256, 256, 0, stream>>>(pBest, pIdx, outRel, outIdx);
}
